// MistralAttention_4612794876097
// MI455X (gfx1250) — matrix operand order repaired and checked by reading, not compiled, not run on hardware
//
#include <hip/hip_runtime.h>

// ---------------------------------------------------------------------------
// MI455X (gfx1250) attention forward.
// B=2, S=2048, H=2048, NH=16, HD=128. wave32; v_wmma_f32_16x16x32_bf16.
// ---------------------------------------------------------------------------

typedef __attribute__((ext_vector_type(16))) __bf16        v16bf;
typedef __attribute__((ext_vector_type(8)))  float         v8f;
typedef __attribute__((ext_vector_type(4)))  float         v4f;
typedef __attribute__((ext_vector_type(4)))  unsigned int  v4u;
typedef __attribute__((ext_vector_type(2)))  unsigned int  v2u;

union Frag {            // 8 VGPRs = 16 bf16 per lane (A or B fragment)
  v4u          q[2];
  unsigned int u[8];
  v16bf        bf;
};

__device__ __forceinline__ unsigned short f2bf(float f) {
  unsigned int u = __float_as_uint(f);
  u += 0x7FFFu + ((u >> 16) & 1u);        // round-to-nearest-even
  return (unsigned short)(u >> 16);
}
__device__ __forceinline__ unsigned int pack2(float a, float b) {
  return (unsigned int)f2bf(a) | ((unsigned int)f2bf(b) << 16);
}

// --------------------------- fp32 -> bf16 convert ---------------------------
__global__ __launch_bounds__(256) void cvt_f32_bf16(
    const float* __restrict__ in, unsigned short* __restrict__ out, int n) {
  int i = (blockIdx.x * blockDim.x + threadIdx.x) * 4;
  if (i + 3 < n) {
    v4f x = *(const v4f*)(in + i);
    v2u r;
    r[0] = pack2(x[0], x[1]);
    r[1] = pack2(x[2], x[3]);
    *(v2u*)(out + i) = r;
  }
}

// ---------------------------------------------------------------------------
// GEMM: computes C^T = W * X^T with W [2048 x 2048] bf16 (rows = out channel),
// X [Ntok x 2048] bf16 (rows = token). Both operands row-contiguous, so every
// WMMA fragment is two b128 loads per lane.
//   mode 0: store natural bf16 [token][chan]   (per-lane b128 vector stores)
//   mode 1: store V^T bf16     [b][chan][seq]  (for attention PV A-operand)
//   mode 2: store natural f32  [token][chan]   (final output)
// Wave tile: 64(M) x 32(N).  Block: 8 waves = 128(M) x 128(N).
// ---------------------------------------------------------------------------
__global__ __launch_bounds__(256) void gemm_wxT(
    const unsigned short* __restrict__ W,
    const unsigned short* __restrict__ X,
    void* __restrict__ OutP,
    int Ntok, int mode) {
  constexpr int Kd = 2048, M = 2048;
  const int lane = threadIdx.x & 31;
  const int lr = lane & 15, hi = lane >> 4;
  const int wave = threadIdx.x >> 5;
  const int mBlocks = M / 128;
  const int bm = blockIdx.x % mBlocks;
  const int bn = blockIdx.x / mBlocks;
  const int mbase = bm * 128 + (wave >> 2) * 64;
  const int nbase = bn * 128 + (wave & 3) * 32;

  v8f acc[4][2] = {};

  for (int kb = 0; kb < Kd; kb += 32) {
    // B fragments (x tokens): lane holds column n = token, K = hi*16 + 0..15
    Frag bfr[2];
#pragma unroll
    for (int ns = 0; ns < 2; ++ns) {
      const unsigned short* xr = X + (size_t)(nbase + ns * 16 + lr) * Kd + kb + hi * 16;
      bfr[ns].q[0] = *(const v4u*)(xr);
      bfr[ns].q[1] = *(const v4u*)(xr + 8);
    }
    // A fragments (W rows): lane holds row m, K = hi*16 + 0..15
#pragma unroll
    for (int ms = 0; ms < 4; ++ms) {
      const unsigned short* wr = W + (size_t)(mbase + ms * 16 + lr) * Kd + kb + hi * 16;
      Frag a;
      a.q[0] = *(const v4u*)(wr);
      a.q[1] = *(const v4u*)(wr + 8);
#pragma unroll
      for (int ns = 0; ns < 2; ++ns)
        acc[ms][ns] = __builtin_amdgcn_wmma_f32_16x16x32_bf16(
            false, a.bf, false, bfr[ns].bf, (short)0, acc[ms][ns], false, false);
    }
  }

  // Epilogue: C^T tile -> lane has token n fixed, 8 consecutive channels.
#pragma unroll
  for (int ms = 0; ms < 4; ++ms) {
#pragma unroll
    for (int ns = 0; ns < 2; ++ns) {
      const int t  = nbase + ns * 16 + lr;       // token
      const int cb = mbase + ms * 16 + 8 * hi;   // first channel of this lane
      v8f c = acc[ms][ns];
      if (mode == 0) {
        unsigned short* O = (unsigned short*)OutP;
        v4u pk;
        pk[0] = pack2(c[0], c[1]); pk[1] = pack2(c[2], c[3]);
        pk[2] = pack2(c[4], c[5]); pk[3] = pack2(c[6], c[7]);
        *(v4u*)(O + (size_t)t * M + cb) = pk;
      } else if (mode == 1) {
        unsigned short* O = (unsigned short*)OutP;  // V^T: [b][chan][seq]
        const int b = t >> 11, s = t & 2047;
#pragma unroll
        for (int v = 0; v < 8; ++v)
          O[((size_t)(b * M + cb + v)) * 2048 + s] = f2bf(c[v]);
      } else {
        float* O = (float*)OutP;
        v4f lo = {c[0], c[1], c[2], c[3]};
        v4f hh = {c[4], c[5], c[6], c[7]};
        *(v4f*)(O + (size_t)t * M + cb) = lo;
        *(v4f*)(O + (size_t)t * M + cb + 4) = hh;
      }
    }
  }
}

// ---------------------------------------------------------------------------
// Flash attention, one wave per (batch, head, 16-query tile).
// S^T = K * Q^T so the score accumulator (lane = query, VGPR = key) matches
// the P^T B-operand of O^T = V^T * P^T up to one shfl_xor(16) half-exchange.
// ---------------------------------------------------------------------------
__global__ __launch_bounds__(256) void attn_fwd(
    const unsigned short* __restrict__ Q,
    const unsigned short* __restrict__ Kmat,
    const unsigned short* __restrict__ Vt,
    const float* __restrict__ mask,
    unsigned short* __restrict__ Aout) {
  constexpr int S = 2048, H = 2048, NH = 16, HD = 128;
  constexpr float SCALE = 0.08838834764831845f;   // 128^-0.5
  constexpr float LOG2E = 1.4426950408889634f;

  const int lane = threadIdx.x & 31;
  const int lr = lane & 15, hi = lane >> 4;
  const int wid = blockIdx.x * 8 + (threadIdx.x >> 5);
  const int b = wid >> 11;                 // 2048 waves per batch
  const int h = (wid >> 7) & (NH - 1);
  const int qbase = (wid & 127) * 16;

  // Q as B-operand of S^T = K*Q^T: column n = query row (contiguous over d).
  Frag qf[4];
  const unsigned short* qr = Q + ((size_t)(b * S + qbase + lr)) * H + h * HD;
#pragma unroll
  for (int ds = 0; ds < 4; ++ds) {
    qf[ds].q[0] = *(const v4u*)(qr + ds * 32 + hi * 8);
    qf[ds].q[1] = *(const v4u*)(qr + ds * 32 + hi * 8 + 16);
  }

  v8f acc[8] = {};                // O^T: 8 d-tiles of [16d x 16q]
  float m_i = -3.0e38f, l_i = 0.f;

  const unsigned short* kp = Kmat + ((size_t)(b * S)) * H + h * HD;
  const unsigned short* vp = Vt + (size_t)b * (size_t)H * S + (size_t)(h * HD) * S;
  const float* mrow = mask + (size_t)(qbase + lr) * S;

  for (int kb = 0; kb < S; kb += 32) {
    v8f s0 = {}, s1 = {};
#pragma unroll
    for (int ds = 0; ds < 4; ++ds) {      // contraction over d = 128
      const unsigned short* kr0 = kp + (size_t)(kb + lr) * H + ds * 32 + hi * 8;
      Frag a0; a0.q[0] = *(const v4u*)(kr0); a0.q[1] = *(const v4u*)(kr0 + 16);
      s0 = __builtin_amdgcn_wmma_f32_16x16x32_bf16(false, a0.bf, false, qf[ds].bf,
                                                   (short)0, s0, false, false);
      const unsigned short* kr1 = kp + (size_t)(kb + 16 + lr) * H + ds * 32 + hi * 8;
      Frag a1; a1.q[0] = *(const v4u*)(kr1); a1.q[1] = *(const v4u*)(kr1 + 16);
      s1 = __builtin_amdgcn_wmma_f32_16x16x32_bf16(false, a1.bf, false, qf[ds].bf,
                                                   (short)0, s1, false, false);
    }

    // scale + additive mask (broadcast over b,h)
    v4f m0a = *(const v4f*)(mrow + kb + 8 * hi);
    v4f m0b = *(const v4f*)(mrow + kb + 8 * hi + 4);
    v4f m1a = *(const v4f*)(mrow + kb + 16 + 8 * hi);
    v4f m1b = *(const v4f*)(mrow + kb + 16 + 8 * hi + 4);
    float p0[8], p1[8];
#pragma unroll
    for (int v = 0; v < 8; ++v) {
      p0[v] = s0[v] * SCALE + (v < 4 ? m0a[v] : m0b[v - 4]);
      p1[v] = s1[v] * SCALE + (v < 4 ? m1a[v] : m1b[v - 4]);
    }

    // online softmax: per-query (per-lane) max over 16 in-lane keys + half-swap
    float mx = p0[0];
#pragma unroll
    for (int v = 1; v < 8; ++v) mx = fmaxf(mx, p0[v]);
#pragma unroll
    for (int v = 0; v < 8; ++v) mx = fmaxf(mx, p1[v]);
    mx = fmaxf(mx, __shfl_xor(mx, 16, 32));
    const float mnew = fmaxf(m_i, mx);
    const float corr = exp2f((m_i - mnew) * LOG2E);
    float rs = 0.f;
#pragma unroll
    for (int v = 0; v < 8; ++v) {
      p0[v] = exp2f((p0[v] - mnew) * LOG2E); rs += p0[v];
      p1[v] = exp2f((p1[v] - mnew) * LOG2E); rs += p1[v];
    }
    rs += __shfl_xor(rs, 16, 32);
    l_i = l_i * corr + rs;
    m_i = mnew;
#pragma unroll
    for (int dt = 0; dt < 8; ++dt) acc[dt] *= corr;

    // Build P^T B-fragment: lane needs keys hi*16 .. hi*16+15 of this block.
    float q0v[8], q1v[8];
#pragma unroll
    for (int v = 0; v < 8; ++v) {
      q0v[v] = __shfl_xor(p0[v], 16, 32);
      q1v[v] = __shfl_xor(p1[v], 16, 32);
    }
    Frag pf;
#pragma unroll
    for (int j = 0; j < 4; ++j) {
      float la = hi ? q1v[2 * j]     : p0[2 * j];
      float lb = hi ? q1v[2 * j + 1] : p0[2 * j + 1];
      float ha = hi ? p1[2 * j]      : q0v[2 * j];
      float hb = hi ? p1[2 * j + 1]  : q0v[2 * j + 1];
      pf.u[j]     = pack2(la, lb);
      pf.u[4 + j] = pack2(ha, hb);
    }

    // O^T += V^T * P^T  (A = V^T rows contiguous over seq)
#pragma unroll
    for (int dt = 0; dt < 8; ++dt) {
      const unsigned short* vr = vp + (size_t)(dt * 16 + lr) * S + kb + hi * 16;
      Frag a; a.q[0] = *(const v4u*)(vr); a.q[1] = *(const v4u*)(vr + 8);
      acc[dt] = __builtin_amdgcn_wmma_f32_16x16x32_bf16(false, a.bf, false, pf.bf,
                                                        (short)0, acc[dt], false, false);
    }
  }

  // Normalize and store O in natural [token][chan] bf16 (8 consecutive chans/lane).
  const float inv = 1.f / l_i;
  unsigned short* op = Aout + ((size_t)(b * S + qbase + lr)) * H + h * HD + 8 * hi;
#pragma unroll
  for (int dt = 0; dt < 8; ++dt) {
    v8f c = acc[dt];
    v4u pk;
    pk[0] = pack2(c[0] * inv, c[1] * inv);
    pk[1] = pack2(c[2] * inv, c[3] * inv);
    pk[2] = pack2(c[4] * inv, c[5] * inv);
    pk[3] = pack2(c[6] * inv, c[7] * inv);
    *(v4u*)(op + dt * 16) = pk;
  }
}

// ---------------------------------------------------------------------------
extern "C" void kernel_launch(void* const* d_in, const int* in_sizes, int n_in,
                              void* d_out, int out_size, void* d_ws, size_t ws_size,
                              hipStream_t stream) {
  const float* x    = (const float*)d_in[0];
  const float* mask = (const float*)d_in[1];
  const float* wq   = (const float*)d_in[2];
  const float* wk   = (const float*)d_in[3];
  const float* wv   = (const float*)d_in[4];
  const float* wo   = (const float*)d_in[5];

  constexpr int H = 2048, Ntok = 2 * 2048;
  constexpr size_t NX = (size_t)Ntok * H;   // activations: 8,388,608 elems
  constexpr size_t NW = (size_t)H * H;      // weights:     4,194,304 elems

  unsigned short* p   = (unsigned short*)d_ws;
  unsigned short* xb  = p; p += NX;
  unsigned short* wqb = p; p += NW;
  unsigned short* wkb = p; p += NW;
  unsigned short* wvb = p; p += NW;
  unsigned short* wob = p; p += NW;
  unsigned short* Qb  = p; p += NX;
  unsigned short* Kb  = p; p += NX;
  unsigned short* Vtb = p; p += NX;
  unsigned short* Ab  = p; p += NX;
  (void)ws_size; (void)in_sizes; (void)n_in; (void)out_size;

  dim3 blk(256);
  cvt_f32_bf16<<<(unsigned)(NX / 1024), blk, 0, stream>>>(x,  xb,  (int)NX);
  cvt_f32_bf16<<<(unsigned)(NW / 1024), blk, 0, stream>>>(wq, wqb, (int)NW);
  cvt_f32_bf16<<<(unsigned)(NW / 1024), blk, 0, stream>>>(wk, wkb, (int)NW);
  cvt_f32_bf16<<<(unsigned)(NW / 1024), blk, 0, stream>>>(wv, wvb, (int)NW);
  cvt_f32_bf16<<<(unsigned)(NW / 1024), blk, 0, stream>>>(wo, wob, (int)NW);

  // (M/128)*(Ntok/128) = 16*32 = 512 blocks
  gemm_wxT<<<512, blk, 0, stream>>>(wqb, xb, Qb,  Ntok, 0);
  gemm_wxT<<<512, blk, 0, stream>>>(wkb, xb, Kb,  Ntok, 0);
  gemm_wxT<<<512, blk, 0, stream>>>(wvb, xb, Vtb, Ntok, 1);

  // 2*16*128 = 4096 waves / 8 waves-per-block = 512 blocks
  attn_fwd<<<512, blk, 0, stream>>>(Qb, Kb, Vtb, mask, Ab);

  gemm_wxT<<<512, blk, 0, stream>>>(wob, Ab, d_out, Ntok, 2);
}
